// Transformer_Block_85976655331539
// MI455X (gfx1250) — compile-verified, matrix operand order checked by reading, not run on hardware
//
#include <hip/hip_runtime.h>
#include <hip/hip_bf16.h>
#include <stdint.h>

// ---------------------------------------------------------------------------
// Transformer block for MI455X (gfx1250, wave32, WMMA bf16 -> f32 accum)
// GEMMs stage tiles through LDS with gfx1250 async loads (ASYNCcnt).
// ---------------------------------------------------------------------------

typedef __attribute__((ext_vector_type(16))) __bf16 v16bf;
typedef __attribute__((ext_vector_type(8)))  float  v8f;

#define EMB      1024
#define HEADS    16
#define HEAD_DIM 64
#define FF       4096
#define SEQ      2048
#define BATCH    4
#define ROWS     (BATCH * SEQ)          // 8192

// GEMM block tile
#define BM 64
#define BN 256
#define BK 32

static __device__ __forceinline__ v8f wmma_bf16(v16bf a, v16bf b, v8f c) {
  return __builtin_amdgcn_wmma_f32_16x16x32_bf16(
      /*neg_a=*/false, a, /*neg_b=*/false, b,
      /*c_mod=*/(short)0, c, /*reuse_a=*/false, /*reuse_b=*/false);
}

// Async copy 16 bytes/lane from global memory into LDS (gfx1250
// GLOBAL_LOAD_ASYNC_TO_LDS_B128, tracked by ASYNCcnt). The LDS address is the
// low 32 bits of the generic pointer (ISA 10.2 aperture rules).
static __device__ __forceinline__ void async_cp16(void* lds, const void* g) {
  asm volatile("global_load_async_to_lds_b128 %0, %1, off"
               :
               : "v"((uint32_t)(size_t)lds), "v"((uint64_t)(size_t)g)
               : "memory");
}
static __device__ __forceinline__ void wait_async0() {
  asm volatile("s_wait_asynccnt 0x0" ::: "memory");
}

// Load a 16x32 bf16 fragment (A layout; B layout is the exact dual with the
// "row" being the output column) from a row-major matrix with leading dim ld.
// ISA 7.12.2: lane holds row m=lane%16, VGPR j holds the contiguous bf16 pair
//   K = k0 + 16*(j/4) + 8*(lane/16) + 2*(j%4).
static __device__ __forceinline__ v16bf load_frag(const __bf16* __restrict__ base,
                                                  int ld, int row0, int k0) {
  const int lane = threadIdx.x & 31;
  const int m = lane & 15, half = lane >> 4;
  union { v16bf v; uint32_t u[8]; } f;
  const __bf16* p = base + (size_t)(row0 + m) * (size_t)ld + k0 + half * 8;
#pragma unroll
  for (int j = 0; j < 8; ++j) {
    const int kk = ((j >> 2) << 4) + ((j & 3) << 1);
    f.u[j] = *(const uint32_t*)(p + kk);
  }
  return f.v;
}

// Same fragment pattern from an LDS tile with ld = 32 (two ds_load_b128 worth).
static __device__ __forceinline__ v16bf load_frag_lds(const __bf16* sp) {
  const int lane = threadIdx.x & 31;
  const int m = lane & 15, half = lane >> 4;
  union { v16bf v; uint32_t u[8]; } f;
  const __bf16* p = sp + m * 32 + half * 8;
#pragma unroll
  for (int j = 0; j < 8; ++j) {
    const int kk = ((j >> 2) << 4) + ((j & 3) << 1);
    f.u[j] = *(const uint32_t*)(p + kk);
  }
  return f.v;
}

// ---------------------------------------------------------------------------
// Weight transpose + cast:  Wt[n*K + k] = (bf16) W[k*N + n]
// ---------------------------------------------------------------------------
__global__ void transpose_cast_kernel(const float* __restrict__ W,
                                      __bf16* __restrict__ Wt, int K, int N) {
  const size_t idx = (size_t)blockIdx.x * 256 + threadIdx.x;
  if (idx >= (size_t)K * N) return;
  const int n = (int)(idx / K);
  const int k = (int)(idx % K);
  Wt[idx] = (__bf16)W[(size_t)k * N + n];
}

// ---------------------------------------------------------------------------
// LayerNorm (one block per row of 1024), f32 in -> bf16 out
// ---------------------------------------------------------------------------
__global__ __launch_bounds__(256)
void layernorm_kernel(const float* __restrict__ x, const float* __restrict__ sc,
                      const float* __restrict__ sh, __bf16* __restrict__ out) {
  const int row = blockIdx.x;
  const float* xr = x + (size_t)row * EMB;
  float v[4], s = 0.f, ss = 0.f;
#pragma unroll
  for (int i = 0; i < 4; ++i) {
    v[i] = xr[threadIdx.x + 256 * i];
    s += v[i];
    ss += v[i] * v[i];
  }
#pragma unroll
  for (int off = 1; off < 32; off <<= 1) {
    s  += __shfl_xor(s, off);
    ss += __shfl_xor(ss, off);
  }
  __shared__ float rs[8], rss[8];
  const int wid = threadIdx.x >> 5;
  if ((threadIdx.x & 31) == 0) { rs[wid] = s; rss[wid] = ss; }
  __syncthreads();
  s = 0.f; ss = 0.f;
#pragma unroll
  for (int i = 0; i < 8; ++i) { s += rs[i]; ss += rss[i]; }
  const float mean = s * (1.f / EMB);
  const float var  = ss * (1.f / EMB) - mean * mean;
  const float rstd = rsqrtf(var + 1e-5f);
#pragma unroll
  for (int i = 0; i < 4; ++i) {
    const int c = threadIdx.x + 256 * i;
    out[(size_t)row * EMB + c] = (__bf16)(sc[c] * ((v[i] - mean) * rstd) + sh[c]);
  }
}

// ---------------------------------------------------------------------------
// WMMA GEMM with async-LDS double-buffered staging:
//   C[M,N] = epilogue(A[M,K] @ Bt[N,K]^T + bias)
// Block: 256 threads = 8 waves (2x4); block tile 64x256; wave tile 32x64.
// Per k-step/wave: 2 A-frags + 4 B-frags from LDS, 8 WMMAs.
// MODE 1: bias + exact-erf GELU -> bf16 row-major
// MODE 2: bias + residual(f32)  -> f32 row-major
// MODE 3: bias -> bf16 scattered to [b,h,t,d]     (Q / K)
// MODE 4: bias -> bf16 scattered to [b,h,d,t]     (V transposed)
// ---------------------------------------------------------------------------
template <int MODE>
__global__ __launch_bounds__(256)
void gemm_wmma_kernel(const __bf16* __restrict__ A, const __bf16* __restrict__ Bt,
                      const float* __restrict__ bias, const float* __restrict__ res,
                      float* __restrict__ outF, __bf16* __restrict__ outB,
                      int M, int N, int K) {
  __shared__ __align__(16) __bf16 As[2][BM * BK];   //  4 KB per buffer
  __shared__ __align__(16) __bf16 Bs[2][BN * BK];   // 16 KB per buffer

  const int tid  = threadIdx.x;
  const int wid  = tid >> 5;
  const int lane = tid & 31;
  const int half = lane >> 4, ln = lane & 15;
  const int mblk = blockIdx.y * BM;
  const int nblk = blockIdx.x * BN;
  const int wm = (wid >> 2) * 32;     // wave row offset in block tile (0/32)
  const int wn = (wid & 3) * 64;      // wave col offset (0/64/128/192)

  // staging geometry: one b128 = 8 bf16; tile row = 32 bf16 = 4 segments
  const int srow = tid >> 2;                 // 0..63
  const int sseg = (tid & 3) * 8;            // 0,8,16,24

  v8f acc[2][4];
#pragma unroll
  for (int i = 0; i < 2; ++i)
#pragma unroll
    for (int j = 0; j < 4; ++j) acc[i][j] = (v8f){};

  const int nk = K / BK;

  // prologue: stage k-tile 0 into buffer 0
  {
    async_cp16(&As[0][srow * BK + sseg], A + (size_t)(mblk + srow) * K + sseg);
#pragma unroll
    for (int i = 0; i < 4; ++i) {
      const int r = srow + i * 64;
      async_cp16(&Bs[0][r * BK + sseg], Bt + (size_t)(nblk + r) * K + sseg);
    }
  }

  for (int kt = 0; kt < nk; ++kt) {
    wait_async0();        // this wave's async copies for tile kt have landed
    __syncthreads();      // => whole tile kt visible to all waves

    if (kt + 1 < nk) {    // overlap: stage tile kt+1 into the other buffer
      const int buf = (kt + 1) & 1;
      const int kk = (kt + 1) * BK;
      async_cp16(&As[buf][srow * BK + sseg],
                 A + (size_t)(mblk + srow) * K + kk + sseg);
#pragma unroll
      for (int i = 0; i < 4; ++i) {
        const int r = srow + i * 64;
        async_cp16(&Bs[buf][r * BK + sseg],
                   Bt + (size_t)(nblk + r) * K + kk + sseg);
      }
    }

    const __bf16* a_t = As[kt & 1];
    const __bf16* b_t = Bs[kt & 1];
    const v16bf af0 = load_frag_lds(a_t + (wm)      * BK);
    const v16bf af1 = load_frag_lds(a_t + (wm + 16) * BK);
#pragma unroll
    for (int j = 0; j < 4; ++j) {
      const v16bf bf = load_frag_lds(b_t + (wn + j * 16) * BK);
      acc[0][j] = wmma_bf16(af0, bf, acc[0][j]);
      acc[1][j] = wmma_bf16(af1, bf, acc[1][j]);
    }
    // no trailing barrier needed: next iteration writes the other buffer, and
    // every wave's ds reads were consumed (dscnt waited) before its next
    // top-of-loop barrier.
  }

  // ---- epilogue ----
#pragma unroll
  for (int i = 0; i < 2; ++i) {
#pragma unroll
    for (int j = 0; j < 4; ++j) {
#pragma unroll
      for (int r = 0; r < 8; ++r) {
        const int row = mblk + wm + i * 16 + r + 8 * half;
        const int col = nblk + wn + j * 16 + ln;
        float v = acc[i][j][r] + bias[col];
        if (MODE == 1) {
          v = 0.5f * v * (1.0f + erff(v * 0.70710678118654752f));
          outB[(size_t)row * N + col] = (__bf16)v;
        } else if (MODE == 2) {
          outF[(size_t)row * N + col] = v + res[(size_t)row * N + col];
        } else if (MODE == 3) {
          const int bb = row >> 11, t = row & (SEQ - 1);
          const int hh = col >> 6,  d = col & (HEAD_DIM - 1);
          outB[(((size_t)(bb * HEADS + hh)) * SEQ + t) * HEAD_DIM + d] = (__bf16)v;
        } else if (MODE == 4) {
          const int bb = row >> 11, t = row & (SEQ - 1);
          const int hh = col >> 6,  d = col & (HEAD_DIM - 1);
          outB[(((size_t)(bb * HEADS + hh)) * HEAD_DIM + d) * SEQ + t] = (__bf16)v;
        } else {
          outB[(size_t)row * N + col] = (__bf16)v;
        }
      }
    }
  }
}

// ---------------------------------------------------------------------------
// Causal flash attention. Grid = (SEQ/128, BATCH*HEADS), 256 threads (8 waves).
// Each wave owns a 16-row q strip; loops over 32-key tiles with online softmax.
// q,k: [b,h,t,64] bf16; vt: [b,h,64,t] bf16; ctx out: [b,t,1024] bf16.
// ---------------------------------------------------------------------------
__global__ __launch_bounds__(256)
void attn_kernel(const __bf16* __restrict__ q, const __bf16* __restrict__ k,
                 const __bf16* __restrict__ vt, __bf16* __restrict__ ctx) {
  const int bh = blockIdx.y;                 // 0..63
  const int b  = bh >> 4, hh = bh & 15;
  const int wid  = threadIdx.x >> 5;
  const int lane = threadIdx.x & 31;
  const int half = lane >> 4, ln = lane & 15;
  const int qr = blockIdx.x * 128 + wid * 16;      // q-row base within sequence

  const __bf16* qp = q  + (size_t)bh * SEQ * HEAD_DIM;
  const __bf16* kp = k  + (size_t)bh * SEQ * HEAD_DIM;
  const __bf16* vp = vt + (size_t)bh * HEAD_DIM * SEQ;

  __shared__ __align__(16) __bf16 psh[8 * 16 * 32];   // per-wave 16x32 P tile
  __bf16* myp = &psh[wid * 512];

  const v16bf qf0 = load_frag(qp, HEAD_DIM, qr, 0);
  const v16bf qf1 = load_frag(qp, HEAD_DIM, qr, 32);

  v8f acc0 = {}, acc1 = {}, acc2 = {}, acc3 = {};
  float mrow[8], lrow[8];
#pragma unroll
  for (int r = 0; r < 8; ++r) { mrow[r] = -3.0e38f; lrow[r] = 0.f; }
  const float scale = 0.125f;   // 1/sqrt(64)

  const int kend = qr + 15;     // causal upper bound for this wave's rows
  for (int kb = 0; kb <= kend; kb += 32) {
    // ---- S = Q @ K^T for 16q x 32keys (two 16x16 tiles, d accumulated) ----
    v8f s0 = {}, s1 = {};
    s0 = wmma_bf16(qf0, load_frag(kp, HEAD_DIM, kb,       0), s0);
    s0 = wmma_bf16(qf1, load_frag(kp, HEAD_DIM, kb,      32), s0);
    s1 = wmma_bf16(qf0, load_frag(kp, HEAD_DIM, kb + 16,  0), s1);
    s1 = wmma_bf16(qf1, load_frag(kp, HEAD_DIM, kb + 16, 32), s1);

    // ---- online softmax per row (rows live across lanes within a half) ----
#pragma unroll
    for (int r = 0; r < 8; ++r) {
      const int row = qr + r + 8 * half;
      const int c0 = kb + ln, c1 = kb + 16 + ln;
      float e0 = (c0 <= row) ? s0[r] * scale : -3.0e38f;
      float e1 = (c1 <= row) ? s1[r] * scale : -3.0e38f;
      float mx = fmaxf(e0, e1);
#pragma unroll
      for (int off = 1; off < 16; off <<= 1) mx = fmaxf(mx, __shfl_xor(mx, off));
      const float mnew = fmaxf(mrow[r], mx);
      const float corr = __expf(mrow[r] - mnew);
      const float p0 = __expf(e0 - mnew);
      const float p1 = __expf(e1 - mnew);
      float ps = p0 + p1;
#pragma unroll
      for (int off = 1; off < 16; off <<= 1) ps += __shfl_xor(ps, off);
      lrow[r] = lrow[r] * corr + ps;
      mrow[r] = mnew;
      acc0[r] *= corr; acc1[r] *= corr; acc2[r] *= corr; acc3[r] *= corr;
      const int m = r + 8 * half;
      myp[m * 32 + ln]      = (__bf16)p0;
      myp[m * 32 + 16 + ln] = (__bf16)p1;
    }
    // order the wave-private LDS P-tile store before the fragment reload
    asm volatile("s_wait_dscnt 0" ::: "memory");
    const v16bf pf = load_frag_lds(myp);

    // ---- ctx += P @ V  (4 d-tiles of 16, K = 32 keys) ----
    acc0 = wmma_bf16(pf, load_frag(vp, SEQ,  0, kb), acc0);
    acc1 = wmma_bf16(pf, load_frag(vp, SEQ, 16, kb), acc1);
    acc2 = wmma_bf16(pf, load_frag(vp, SEQ, 32, kb), acc2);
    acc3 = wmma_bf16(pf, load_frag(vp, SEQ, 48, kb), acc3);
  }

  // ---- normalize and scatter into head-concat row-major ctx [b,t,1024] ----
  v8f* accs[4] = { &acc0, &acc1, &acc2, &acc3 };
#pragma unroll
  for (int n = 0; n < 4; ++n) {
#pragma unroll
    for (int r = 0; r < 8; ++r) {
      const int row = qr + r + 8 * half;
      const int col = hh * HEAD_DIM + n * 16 + ln;
      const float o = (*accs[n])[r] / lrow[r];
      ctx[((size_t)b * SEQ + row) * EMB + col] = (__bf16)o;
    }
  }
}

// ---------------------------------------------------------------------------
// Orchestration
// ---------------------------------------------------------------------------
extern "C" void kernel_launch(void* const* d_in, const int* in_sizes, int n_in,
                              void* d_out, int out_size, void* d_ws, size_t ws_size,
                              hipStream_t stream) {
  (void)in_sizes; (void)n_in; (void)out_size; (void)ws_size;
  const float* x    = (const float*)d_in[0];
  const float* ln1s = (const float*)d_in[1];
  const float* ln1b = (const float*)d_in[2];
  const float* ln2s = (const float*)d_in[3];
  const float* ln2b = (const float*)d_in[4];
  const float* wq = (const float*)d_in[5];
  const float* bq = (const float*)d_in[6];
  const float* wk = (const float*)d_in[7];
  const float* bk = (const float*)d_in[8];
  const float* wv = (const float*)d_in[9];
  const float* bv = (const float*)d_in[10];
  const float* wo = (const float*)d_in[11];
  const float* bo = (const float*)d_in[12];
  const float* w1 = (const float*)d_in[13];
  const float* b1 = (const float*)d_in[14];
  const float* w2 = (const float*)d_in[15];
  const float* b2 = (const float*)d_in[16];

  // ---- workspace bump allocator (256B aligned) ----
  char* wsp = (char*)d_ws;
  auto alloc = [&](size_t bytes) -> char* {
    char* p = wsp;
    wsp += (bytes + 255) & ~(size_t)255;
    return p;
  };
  __bf16* wqT = (__bf16*)alloc((size_t)EMB * EMB * 2);
  __bf16* wkT = (__bf16*)alloc((size_t)EMB * EMB * 2);
  __bf16* wvT = (__bf16*)alloc((size_t)EMB * EMB * 2);
  __bf16* woT = (__bf16*)alloc((size_t)EMB * EMB * 2);
  __bf16* w1T = (__bf16*)alloc((size_t)EMB * FF * 2);
  __bf16* w2T = (__bf16*)alloc((size_t)EMB * FF * 2);
  __bf16* hbuf  = (__bf16*)alloc((size_t)ROWS * EMB * 2);
  __bf16* qbuf  = (__bf16*)alloc((size_t)ROWS * EMB * 2);
  __bf16* kbuf  = (__bf16*)alloc((size_t)ROWS * EMB * 2);
  __bf16* vtbuf = (__bf16*)alloc((size_t)ROWS * EMB * 2);
  __bf16* ctxb  = (__bf16*)alloc((size_t)ROWS * EMB * 2);
  float*  x1    = (float*)alloc((size_t)ROWS * EMB * 4);
  __bf16* h2    = (__bf16*)alloc((size_t)ROWS * EMB * 2);
  __bf16* ffn   = (__bf16*)alloc((size_t)ROWS * FF * 2);

  // ---- weight transpose + bf16 cast ----
  {
    const int nE = EMB * EMB, nF = EMB * FF;
    transpose_cast_kernel<<<(nE + 255) / 256, 256, 0, stream>>>(wq, wqT, EMB, EMB);
    transpose_cast_kernel<<<(nE + 255) / 256, 256, 0, stream>>>(wk, wkT, EMB, EMB);
    transpose_cast_kernel<<<(nE + 255) / 256, 256, 0, stream>>>(wv, wvT, EMB, EMB);
    transpose_cast_kernel<<<(nE + 255) / 256, 256, 0, stream>>>(wo, woT, EMB, EMB);
    transpose_cast_kernel<<<(nF + 255) / 256, 256, 0, stream>>>(w1, w1T, EMB, FF);
    transpose_cast_kernel<<<(nF + 255) / 256, 256, 0, stream>>>(w2, w2T, FF, EMB);
  }

  // ---- attention sublayer ----
  layernorm_kernel<<<ROWS, 256, 0, stream>>>(x, ln1s, ln1b, hbuf);

  dim3 gE(EMB / BN, ROWS / BM);    // (4, 128)
  gemm_wmma_kernel<3><<<gE, 256, 0, stream>>>(hbuf, wqT, bq, nullptr, nullptr, qbuf,
                                              ROWS, EMB, EMB);
  gemm_wmma_kernel<3><<<gE, 256, 0, stream>>>(hbuf, wkT, bk, nullptr, nullptr, kbuf,
                                              ROWS, EMB, EMB);
  gemm_wmma_kernel<4><<<gE, 256, 0, stream>>>(hbuf, wvT, bv, nullptr, nullptr, vtbuf,
                                              ROWS, EMB, EMB);

  attn_kernel<<<dim3(SEQ / 128, BATCH * HEADS), 256, 0, stream>>>(qbuf, kbuf, vtbuf, ctxb);

  gemm_wmma_kernel<2><<<gE, 256, 0, stream>>>(ctxb, woT, bo, x, x1, nullptr,
                                              ROWS, EMB, EMB);

  // ---- MLP sublayer ----
  layernorm_kernel<<<ROWS, 256, 0, stream>>>(x1, ln2s, ln2b, h2);

  dim3 gF(FF / BN, ROWS / BM);     // (16, 128)
  gemm_wmma_kernel<1><<<gF, 256, 0, stream>>>(h2, w1T, b1, nullptr, nullptr, ffn,
                                              ROWS, FF, EMB);
  gemm_wmma_kernel<2><<<gE, 256, 0, stream>>>(ffn, w2T, b2, x1, (float*)d_out, nullptr,
                                              ROWS, EMB, FF);
}
